// GATConv_25864293056852
// MI455X (gfx1250) — compile-verified, matrix operand order checked by reading, not run on hardware
//
#include <hip/hip_runtime.h>

// GAT layer, MI455X (gfx1250, wave32, WMMA f16 16x16x32).
// b=8, N=2048, Din=256, H=4, F=64 (H*F=256).
// Inputs (f32): adj[8,2048,2048], x[8,2048,256], W[256,256], a_src[4,64],
//               a_dst[4,64], bias[256].  Output (f32): [8,2048,256].
//
// Design: fused single-pass attention. shift_i = lrelu(s_src_i + max_j s_dst_j)
// upper-bounds every masked-in score (lrelu monotone), so softmax needs no
// online rescale: acc += P@V with P = exp(score-shift), divide by rowsum at end.
// adj (134 MB, the dominant HBM traffic) is read exactly once: one wave owns a
// 16-row i-tile and all 4 heads.

typedef __attribute__((ext_vector_type(16))) _Float16 v16h;
typedef __attribute__((ext_vector_type(8)))  _Float16 v8h;
typedef __attribute__((ext_vector_type(8)))  float    v8f;
typedef __attribute__((ext_vector_type(4)))  float    v4f;

#define NEG_SLOPE 0.2f
#define THRED     0.5f

#define NB   8
#define NN   2048
#define DIN  256
#define NH   4
#define NF   64
#define DOUT 256   // NH*NF

// ---------------- Kernel 1: W (f32, [K][N]) -> Wt (f16, [N][K]) -------------
__global__ __launch_bounds__(256) void cvt_w(const float* __restrict__ W,
                                             _Float16* __restrict__ Wt) {
  int idx = blockIdx.x * 256 + threadIdx.x;       // 65536 elements
  int k = idx >> 8, n = idx & 255;
  Wt[n * DIN + k] = (_Float16)W[idx];
}

// ---------------- Kernel 2: hT = (x @ W)^T in f16 ---------------------------
// hT layout: [b][DOUT][NN] (feature-major, node contiguous).
// One wave computes a 16-row x 64-col tile. 4096 waves total.
__global__ __launch_bounds__(128) void gemm_xw(const float* __restrict__ x,
                                               const _Float16* __restrict__ Wt,
                                               _Float16* __restrict__ hT) {
  const int lane  = threadIdx.x & 31;
  const int w     = (blockIdx.x * blockDim.x + threadIdx.x) >> 5; // 0..4095
  const int itile = w >> 2;      // 0..1023 (16 rows each over b*N=16384)
  const int cq    = w & 3;       // column quad: n0 = cq*64
  const int rl    = lane & 15;
  const int kb    = (lane >> 4) * 8;

  const float* xrow = x + (size_t)(itile * 16 + rl) * DIN;

  v8f acc0 = {}, acc1 = {}, acc2 = {}, acc3 = {};
#pragma unroll
  for (int k0 = 0; k0 < DIN; k0 += 32) {
    // A fragment: row rl, K = k0+kb+{0..7}, k0+kb+16+{0..7}
    float fa[16];
    *(v4f*)(fa + 0)  = *(const v4f*)(xrow + k0 + kb + 0);
    *(v4f*)(fa + 4)  = *(const v4f*)(xrow + k0 + kb + 4);
    *(v4f*)(fa + 8)  = *(const v4f*)(xrow + k0 + kb + 16);
    *(v4f*)(fa + 12) = *(const v4f*)(xrow + k0 + kb + 20);
    v16h a;
#pragma unroll
    for (int t = 0; t < 16; ++t) a[t] = (_Float16)fa[t];

#pragma unroll
    for (int c = 0; c < 4; ++c) {
      const int col = cq * 64 + c * 16 + rl;              // output feature
      const _Float16* bp = Wt + (size_t)col * DIN + k0 + kb;
      v8h b0 = *(const v8h*)bp;
      v8h b1 = *(const v8h*)(bp + 16);
      v16h bf;
#pragma unroll
      for (int t = 0; t < 8; ++t) { bf[t] = b0[t]; bf[t + 8] = b1[t]; }
      v8f& acc = (c == 0) ? acc0 : (c == 1) ? acc1 : (c == 2) ? acc2 : acc3;
      acc = __builtin_amdgcn_wmma_f32_16x16x32_f16(false, a, false, bf,
                                                   (short)0, acc, false, false);
    }
  }

  // Store transposed: hT[b][col][node]
  const int mbase = (lane >> 4) * 8;
  const int node0 = itile * 16;
  const int bb    = node0 >> 11;           // node0 / 2048 (tile never crosses b)
  const int nb0   = node0 & 2047;
#pragma unroll
  for (int c = 0; c < 4; ++c) {
    const int col = cq * 64 + c * 16 + rl;
    v8f& acc = (c == 0) ? acc0 : (c == 1) ? acc1 : (c == 2) ? acc2 : acc3;
    _Float16* dst = hT + ((size_t)bb * DOUT + col) * NN + nb0 + mbase;
#pragma unroll
    for (int r = 0; r < 8; ++r) dst[r] = (_Float16)acc[r];
  }
}

// ---------------- Kernel 3: s_src/s_dst + per-(b,head) max of s_dst ---------
__global__ __launch_bounds__(256) void proj(const _Float16* __restrict__ hT,
                                            const float* __restrict__ a_src,
                                            const float* __restrict__ a_dst,
                                            float* __restrict__ s_src,
                                            float* __restrict__ s_dst,
                                            float* __restrict__ maxdst) {
  const int bh = blockIdx.x;               // 0..31 == b*4+h
  const int h  = bh & 3;
  const _Float16* base = hT + ((size_t)(bh >> 2) * DOUT + h * NF) * NN;
  __shared__ float red[256];
  float lmax = -3.0e38f;
#pragma unroll
  for (int i = 0; i < 8; ++i) {
    const int n = threadIdx.x + i * 256;
    float ss = 0.f, sd = 0.f;
#pragma unroll
    for (int f = 0; f < NF; ++f) {
      const float hv = (float)base[(size_t)f * NN + n];
      ss += hv * a_src[h * NF + f];
      sd += hv * a_dst[h * NF + f];
    }
    s_src[(size_t)bh * NN + n] = ss;
    s_dst[(size_t)bh * NN + n] = sd;
    lmax = fmaxf(lmax, sd);
  }
  red[threadIdx.x] = lmax;
  __syncthreads();
  for (int s = 128; s > 0; s >>= 1) {
    if (threadIdx.x < s) red[threadIdx.x] = fmaxf(red[threadIdx.x], red[threadIdx.x + s]);
    __syncthreads();
  }
  if (threadIdx.x == 0) maxdst[bh] = red[0];
}

// ---------------- Kernel 4: fused masked softmax + P@V + residual + ELU -----
// One wave per (b, 16-row i-tile); all 4 heads; adj read once.
__global__ __launch_bounds__(128) void attn(const float* __restrict__ adj,
                                            const float* __restrict__ x,
                                            const _Float16* __restrict__ hT,
                                            const float* __restrict__ s_src,
                                            const float* __restrict__ s_dst,
                                            const float* __restrict__ maxdst,
                                            const float* __restrict__ bias,
                                            float* __restrict__ out) {
  const int lane  = threadIdx.x & 31;
  const int w     = (blockIdx.x * blockDim.x + threadIdx.x) >> 5; // 0..1023
  const int b     = w >> 7;        // batch
  const int itile = w & 127;
  const int i0    = itile * 16;
  const int rl    = lane & 15;
  const int kb    = (lane >> 4) * 8;

  // Per-head row scalars: s_src of this lane's row, and the softmax shift.
  float ssrc[NH], shiftv[NH];
#pragma unroll
  for (int h = 0; h < NH; ++h) {
    const float s = s_src[(size_t)(b * NH + h) * NN + i0 + rl];
    ssrc[h] = s;
    const float m = s + maxdst[b * NH + h];
    shiftv[h] = (m > 0.f) ? m : NEG_SLOPE * m;   // lrelu(s_src_i + max_j s_dst_j)
  }

  v8f acc[NH][4];
#pragma unroll
  for (int h = 0; h < NH; ++h)
#pragma unroll
    for (int c = 0; c < 4; ++c) acc[h][c] = (v8f){};
  float rowsum[NH] = {0.f, 0.f, 0.f, 0.f};

  const float*    adjrow = adj + ((size_t)b * NN + i0 + rl) * NN;
  const _Float16* hTb    = hT + (size_t)b * DOUT * NN;
  const float*    sdb    = s_dst + (size_t)b * NH * NN;

  for (int j0 = 0; j0 < NN; j0 += 32) {
    // adj for this lane's row, K = j0+kb+{0..7,16..23}
    float av[16];
    *(v4f*)(av + 0)  = *(const v4f*)(adjrow + j0 + kb + 0);
    *(v4f*)(av + 4)  = *(const v4f*)(adjrow + j0 + kb + 4);
    *(v4f*)(av + 8)  = *(const v4f*)(adjrow + j0 + kb + 16);
    *(v4f*)(av + 12) = *(const v4f*)(adjrow + j0 + kb + 20);

#pragma unroll
    for (int h = 0; h < NH; ++h) {
      float sv[16];
      const float* sdp = sdb + (size_t)h * NN + j0 + kb;
      *(v4f*)(sv + 0)  = *(const v4f*)(sdp + 0);
      *(v4f*)(sv + 4)  = *(const v4f*)(sdp + 4);
      *(v4f*)(sv + 8)  = *(const v4f*)(sdp + 16);
      *(v4f*)(sv + 12) = *(const v4f*)(sdp + 20);

      v16h pa;
      float rs = 0.f;
#pragma unroll
      for (int t = 0; t < 16; ++t) {
        float sc = ssrc[h] + sv[t];
        sc = (sc > 0.f) ? sc : NEG_SLOPE * sc;            // leakyReLU
        const float e = (av[t] < THRED) ? 0.f : __expf(sc - shiftv[h]);
        rs += e;
        pa[t] = (_Float16)e;
      }
      rowsum[h] += rs;

#pragma unroll
      for (int c = 0; c < 4; ++c) {
        // V fragment: col f = h*64 + c*16 + rl, K contiguous in hT
        const _Float16* bp = hTb + (size_t)(h * NF + c * 16 + rl) * NN + j0 + kb;
        v8h b0 = *(const v8h*)bp;
        v8h b1 = *(const v8h*)(bp + 16);
        v16h bf;
#pragma unroll
        for (int t = 0; t < 8; ++t) { bf[t] = b0[t]; bf[t + 8] = b1[t]; }
        acc[h][c] = __builtin_amdgcn_wmma_f32_16x16x32_f16(
            false, pa, false, bf, (short)0, acc[h][c], false, false);
      }
    }
  }

  // Complete row sums: lane l and l+16 hold halves of row (l&15).
#pragma unroll
  for (int h = 0; h < NH; ++h) rowsum[h] += __shfl_xor(rowsum[h], 16, 32);

  const int mbase = (lane >> 4) * 8;
#pragma unroll
  for (int h = 0; h < NH; ++h) {
#pragma unroll
    for (int c = 0; c < 4; ++c) {
      const int col = h * NF + c * 16 + rl;
      const float bcol = bias[col];
#pragma unroll
      for (int r = 0; r < 8; ++r) {
        const int   node  = i0 + mbase + r;
        const float denom = __shfl(rowsum[h], mbase + r, 32); // full sum row M
        const size_t oidx = ((size_t)b * NN + node) * DOUT + col;
        float v = acc[h][c][r] / denom + x[oidx] + bcol;      // + residual + bias
        v = (v > 0.f) ? v : (__expf(v) - 1.0f);               // ELU (alpha=1)
        out[oidx] = v;
      }
    }
  }
}

// ---------------------------------------------------------------------------
extern "C" void kernel_launch(void* const* d_in, const int* in_sizes, int n_in,
                              void* d_out, int out_size, void* d_ws, size_t ws_size,
                              hipStream_t stream) {
  const float* adj   = (const float*)d_in[0];
  const float* x     = (const float*)d_in[1];
  const float* W     = (const float*)d_in[2];
  const float* a_src = (const float*)d_in[3];
  const float* a_dst = (const float*)d_in[4];
  const float* bias  = (const float*)d_in[5];
  float*       out   = (float*)d_out;

  // Workspace layout (~8.9 MB)
  char* ws = (char*)d_ws;
  _Float16* Wt     = (_Float16*)ws;                                  // 128 KB
  _Float16* hT     = (_Float16*)(ws + 131072);                       // 8 MB
  float*    s_src  = (float*)(ws + 131072 + 8388608);                // 256 KB
  float*    s_dst  = (float*)(ws + 131072 + 8388608 + 262144);       // 256 KB
  float*    maxdst = (float*)(ws + 131072 + 8388608 + 524288);       // 128 B

  cvt_w<<<256, 256, 0, stream>>>(W, Wt);
  gemm_xw<<<1024, 128, 0, stream>>>(x, Wt, hT);          // 4096 waves
  proj<<<NB * NH, 256, 0, stream>>>(hT, a_src, a_dst, s_src, s_dst, maxdst);
  attn<<<256, 128, 0, stream>>>(adj, x, hT, s_src, s_dst, maxdst, bias, out); // 1024 waves
}
